// PairformerStack_60516089201091
// MI455X (gfx1250) — matrix operand order repaired and checked by reading, not compiled, not run on hardware
//
#include <hip/hip_runtime.h>
#include <hip/hip_bf16.h>

// ============================================================================
// Pairformer stack for MI455X (gfx1250, wave32, WMMA).
//
// bf16 WMMA inputs, f32 accumulate (v_wmma_f32_16x16x32_bf16). Weights are
// pre-transposed to NT layout [N][K] so every WMMA operand is loaded as two
// contiguous global_load_b128 per lane. The generic GEMM requires N%64==0 and
// is fully branch-free; N=16 bias projections use a dedicated kernel.
// Working set is L2-resident (192 MB) -> matrix-core bound.
// ============================================================================

typedef __bf16 bf16;
typedef __attribute__((ext_vector_type(16))) __bf16 v16bf;
typedef __attribute__((ext_vector_type(8)))  __bf16 v8bf;
typedef __attribute__((ext_vector_type(8)))  float  v8f;
typedef __attribute__((ext_vector_type(4)))  float  v4f;

#define NRES 256
#define NN   65536   /* 256*256 */
#define CZ   128
#define CS   384

static __device__ __forceinline__ float sigmoidf_(float x) {
    return 1.0f / (1.0f + __expf(-x));
}

static __device__ __forceinline__ v8f v8f_zero() {
    v8f z;
#pragma unroll
    for (int i = 0; i < 8; ++i) z[i] = 0.0f;
    return z;
}

static __device__ __forceinline__ v8f wmma_bf16(v16bf a, v16bf b, v8f c) {
    return __builtin_amdgcn_wmma_f32_16x16x32_bf16(false, a, false, b, (short)0, c,
                                                   false, false);
}

// ---------------------------------------------------------------------------
// Fragment loaders (ISA 7.12.2, wave32), fully vectorized.
//
// A frag (16x32): lane l holds row M=l&15; element e -> K = kb+e,
// kb=(l>>4)*16. One contiguous 16xbf16 run.
// B frag (32x16): lane l holds col N=l&15; element e -> K = (l>>4)*16+e.
// One contiguous 16xbf16 run.
// C/D frag: element r -> row M = r + ((l>>4)<<3), col N = l&15.
// ---------------------------------------------------------------------------

static __device__ __forceinline__ v16bf frag_row(const bf16* __restrict__ base,
                                                 long rs, int row0, int k0, int lane) {
    const bf16* p = base + (long)(row0 + (lane & 15)) * rs + k0 + ((lane >> 4) << 4);
    const v8bf lo = *(const v8bf*)p;          // K = kb .. kb+7 (kb=(l>>4)*16)
    const v8bf hi = *(const v8bf*)(p + 8);    // K = kb+8 .. kb+15
    v16bf f;
#pragma unroll
    for (int e = 0; e < 8; ++e) { f[e] = lo[e]; f[e + 8] = hi[e]; }
    return f;
}

// A frag from f32 source (LDS score strips); rs must keep 16B alignment.
static __device__ __forceinline__ v16bf frag_row_f32(const float* __restrict__ base,
                                                     long rs, int row0, int k0,
                                                     int lane) {
    const float* p = base + (long)(row0 + (lane & 15)) * rs + k0 + ((lane >> 4) << 4);
    const v4f a0 = *(const v4f*)p;
    const v4f a1 = *(const v4f*)(p + 4);
    const v4f b0 = *(const v4f*)(p + 8);
    const v4f b1 = *(const v4f*)(p + 12);
    v16bf f;
#pragma unroll
    for (int e = 0; e < 4; ++e) {
        f[e]      = (bf16)a0[e];
        f[e + 4]  = (bf16)a1[e];
        f[e + 8]  = (bf16)b0[e];
        f[e + 12] = (bf16)b1[e];
    }
    return f;
}

// B frag from NT storage (row-major [rows, rs], contraction contiguous).
static __device__ __forceinline__ v16bf frag_nt(const bf16* __restrict__ base,
                                                long rs, int row0, int k0, int lane) {
    const bf16* p = base + (long)(row0 + (lane & 15)) * rs + k0 + ((lane >> 4) << 4);
    const v8bf lo = *(const v8bf*)p;
    const v8bf hi = *(const v8bf*)(p + 8);
    v16bf f;
#pragma unroll
    for (int e = 0; e < 8; ++e) { f[e] = lo[e]; f[e + 8] = hi[e]; }
    return f;
}

// B frag from KxN storage (strided along k), branchless. Attention V only.
static __device__ __forceinline__ v16bf frag_col_strided(const bf16* __restrict__ base,
                                                         long ldb, int k0, int n0,
                                                         int lane) {
    const bf16* p = base + (long)(k0 + ((lane >> 4) << 4)) * ldb + n0 + (lane & 15);
    v16bf f;
#pragma unroll
    for (int e = 0; e < 16; ++e) f[e] = p[(long)e * ldb];
    return f;
}

// ---------------------------------------------------------------------------
// Generic WMMA GEMM: out[M,N] = A[M,K](bf16) @ W[K,N] with W stored NT as
// Wt[N][K] bf16. REQUIRES N % 64 == 0. Branch-free inner loop.
// mode 0: acc (+bias[n])   mode 1: res+acc   mode 2: res+sigmoid(gate)*acc
// mode 3: sigmoid(acc).   If outb != null, store bf16, else f32 to out.
// Block: 8 waves, wave = 16x64 -> block tile 128x64.
// ---------------------------------------------------------------------------
__global__ void __launch_bounds__(256)
gemm_bf16_kernel(const bf16* __restrict__ A, const bf16* __restrict__ Wt,
                 const float* __restrict__ bias, const float* __restrict__ res,
                 const float* __restrict__ gate, float* __restrict__ out,
                 bf16* __restrict__ outb, int M, int N, int K, int mode) {
    const int lane = threadIdx.x & 31;
    const int wave = threadIdx.x >> 5;
    const int m0 = blockIdx.x * 128 + wave * 16;
    const int n0 = blockIdx.y * 64;
    if (m0 >= M) return;

    __builtin_prefetch(A + (long)m0 * K, 0, 3);  // global_prefetch_b8

    v8f acc[4];
#pragma unroll
    for (int t = 0; t < 4; ++t) acc[t] = v8f_zero();

    for (int k0 = 0; k0 < K; k0 += 32) {
        const v16bf a = frag_row(A, K, m0, k0, lane);
#pragma unroll
        for (int t = 0; t < 4; ++t) {
            const v16bf b = frag_nt(Wt, K, n0 + t * 16, k0, lane);
            acc[t] = wmma_bf16(a, b, acc[t]);
        }
    }

    const int rbase = (lane >> 4) << 3;
    const int col = lane & 15;
#pragma unroll
    for (int t = 0; t < 4; ++t) {
#pragma unroll
        for (int r = 0; r < 8; ++r) {
            const int m = m0 + rbase + r;
            const int n = n0 + t * 16 + col;
            float v = acc[t][r];
            if (bias) v += bias[n];
            const long idx = (long)m * N + n;
            if (mode == 1)      v = res[idx] + v;
            else if (mode == 2) v = res[idx] + sigmoidf_(gate[idx]) * v;
            else if (mode == 3) v = sigmoidf_(v);
            if (outb) outb[idx] = (bf16)v;
            else      out[idx] = v;
        }
    }
}

// ---------------------------------------------------------------------------
// Narrow GEMM for N=16 (attention-bias projections): out[M,16] = A @ Wt[16][K].
// One accumulator per wave; no epilogue options.
// ---------------------------------------------------------------------------
__global__ void __launch_bounds__(256)
gemm_n16_kernel(const bf16* __restrict__ A, const bf16* __restrict__ Wt,
                float* __restrict__ out, int M, int K) {
    const int lane = threadIdx.x & 31;
    const int wave = threadIdx.x >> 5;
    const int m0 = blockIdx.x * 128 + wave * 16;
    if (m0 >= M) return;

    v8f acc = v8f_zero();
    for (int k0 = 0; k0 < K; k0 += 32) {
        const v16bf a = frag_row(A, K, m0, k0, lane);
        const v16bf b = frag_nt(Wt, K, 0, k0, lane);
        acc = wmma_bf16(a, b, acc);
    }

    const int rbase = (lane >> 4) << 3;
    const int col = lane & 15;
#pragma unroll
    for (int r = 0; r < 8; ++r)
        out[(long)(m0 + rbase + r) * 16 + col] = acc[r];
}

// ---------------------------------------------------------------------------
// Triangle einsum, batched over d: out[i,j,d] = sum_k at[d][i][k]*bt[d][j][k].
// NT GEMM per channel d; at/bt d-major [128][256][256] bf16.
// grid (4,4,128), block = 4 waves, wave = 32x32 tile.
// ---------------------------------------------------------------------------
__global__ void __launch_bounds__(128)
tri_einsum_kernel(const bf16* __restrict__ at, const bf16* __restrict__ bt,
                  float* __restrict__ out) {
    const int lane = threadIdx.x & 31;
    const int wave = threadIdx.x >> 5;
    const int d = blockIdx.z;
    const int m0 = blockIdx.x * 64 + (wave >> 1) * 32;
    const int n0 = blockIdx.y * 64 + (wave & 1) * 32;
    const bf16* Ad = at + (long)d * NN;
    const bf16* Bd = bt + (long)d * NN;

    v8f acc[4];
#pragma unroll
    for (int t = 0; t < 4; ++t) acc[t] = v8f_zero();

    for (int k0 = 0; k0 < 256; k0 += 32) {
        const v16bf a0 = frag_row(Ad, 256, m0, k0, lane);
        const v16bf a1 = frag_row(Ad, 256, m0 + 16, k0, lane);
        const v16bf b0 = frag_nt(Bd, 256, n0, k0, lane);
        const v16bf b1 = frag_nt(Bd, 256, n0 + 16, k0, lane);
        acc[0] = wmma_bf16(a0, b0, acc[0]);
        acc[1] = wmma_bf16(a0, b1, acc[1]);
        acc[2] = wmma_bf16(a1, b0, acc[2]);
        acc[3] = wmma_bf16(a1, b1, acc[3]);
    }

    const int rbase = (lane >> 4) << 3;
    const int col = lane & 15;
#pragma unroll
    for (int t = 0; t < 4; ++t) {
#pragma unroll
        for (int r = 0; r < 8; ++r) {
            const int m = m0 + (t >> 1) * 16 + rbase + r;
            const int n = n0 + (t & 1) * 16 + col;
            out[((long)m * 256 + n) * CZ + d] = acc[t][r];
        }
    }
}

// ---------------------------------------------------------------------------
// Triangle attention (start & end variants via strides).
// Block = (fixed index, 2 strips of 16 rows), grid (256, 8, 4 heads).
// Phase A: S = Q K^T * scale + bias (one 16x16x32 WMMA per tile, d=32).
// Phase B: wave-local softmax over k=256 via wave32 shuffles.
// Phase C: O = P V (16 WMMAs: 8 k-steps x 2 tiles), gate + 1/rowsum, bf16 store.
// Bias buffer is [nn,16] (padded head dim).
// ---------------------------------------------------------------------------
__global__ void __launch_bounds__(64)
tri_attn_kernel(const bf16* __restrict__ q, const bf16* __restrict__ k,
                const bf16* __restrict__ v, const float* __restrict__ bias,
                const float* __restrict__ gsig, bf16* __restrict__ o,
                int starting) {
    __shared__ float S[2][16][272];   // 16B-aligned rows for vector loads
    __shared__ float inv[2][16];

    const int lane = threadIdx.x & 31;
    const int wave = threadIdx.x >> 5;
    const int fixed = blockIdx.x;
    const int h = blockIdx.z;
    const int strip = blockIdx.y * 2 + wave;
    const int j0 = strip * 16;

    long base, rs, b_rs, b_cs;
    if (starting) {  // fixed = i; rows = j; attend over k along row i
        base = (long)fixed * NRES * CZ + h * 32; rs = CZ;
        b_rs = (long)NRES * 16; b_cs = 16;       // bias[j,k,h] in [nn,16]
    } else {         // fixed = j; rows = i; attend over k along column j
        base = (long)fixed * CZ + h * 32; rs = (long)NRES * CZ;
        b_rs = 16; b_cs = (long)NRES * 16;       // bias[k,i,h]
    }
    const float scale = 0.17677669529663687f;    // 1/sqrt(32)
    const int rbase = (lane >> 4) << 3;
    const int col = lane & 15;

    // Phase A ---------------------------------------------------------------
    const v16bf qf = frag_row(q + base, rs, j0, 0, lane);
    for (int kt = 0; kt < 16; ++kt) {
        const v16bf kf = frag_nt(k + base, rs, kt * 16, 0, lane);
        v8f c = v8f_zero();
        c = wmma_bf16(qf, kf, c);
#pragma unroll
        for (int r = 0; r < 8; ++r) {
            const int rr = rbase + r;
            const int cc = kt * 16 + col;
            S[wave][rr][cc] =
                c[r] * scale + bias[(long)h + (long)(j0 + rr) * b_rs + (long)cc * b_cs];
        }
    }
    __syncthreads();

    // Phase B: softmax (2 lanes per row) -------------------------------------
    {
        const int row = lane >> 1, half = lane & 1;
        float mx = -3.4e38f;
        for (int c = half * 128; c < half * 128 + 128; ++c)
            mx = fmaxf(mx, S[wave][row][c]);
        mx = fmaxf(mx, __shfl_xor(mx, 1, 32));
        float sm = 0.0f;
        for (int c = half * 128; c < half * 128 + 128; ++c) {
            const float e = __expf(S[wave][row][c] - mx);
            S[wave][row][c] = e;
            sm += e;
        }
        sm += __shfl_xor(sm, 1, 32);
        if (half == 0) inv[wave][row] = 1.0f / sm;
    }
    __syncthreads();

    // Phase C: O = P V -------------------------------------------------------
    v8f o0 = v8f_zero(), o1 = v8f_zero();
    for (int k0 = 0; k0 < 256; k0 += 32) {
        const v16bf pf = frag_row_f32(&S[wave][0][0], 272, 0, k0, lane);
        const v16bf v0 = frag_col_strided(v + base, rs, k0, 0, lane);
        const v16bf v1 = frag_col_strided(v + base, rs, k0, 16, lane);
        o0 = wmma_bf16(pf, v0, o0);
        o1 = wmma_bf16(pf, v1, o1);
    }
#pragma unroll
    for (int r = 0; r < 8; ++r) {
        const int rr = rbase + r;
        const int x = j0 + rr;
        const float sc = inv[wave][rr];
        const long i0 = base + (long)x * rs + col;
        const long i1 = i0 + 16;
        o[i0] = (bf16)(o0[r] * sc * gsig[i0]);
        o[i1] = (bf16)(o1[r] * sc * gsig[i1]);
    }
}

// ---------------------------------------------------------------------------
// LayerNorm over last dim C; one block (128 threads) per row; bf16 output.
// ---------------------------------------------------------------------------
__global__ void __launch_bounds__(128)
ln_kernel(const float* __restrict__ x, const float* __restrict__ gw,
          const float* __restrict__ bw, bf16* __restrict__ out, int C) {
    const int row = blockIdx.x;
    const float* xr = x + (long)row * C;
    bf16* orow = out + (long)row * C;
    const int t = threadIdx.x;
    __shared__ float red[128];

    float s = 0.0f;
    for (int c = t; c < C; c += 128) s += xr[c];
    red[t] = s;
    __syncthreads();
    for (int off = 64; off > 0; off >>= 1) {
        if (t < off) red[t] += red[t + off];
        __syncthreads();
    }
    const float mean = red[0] / (float)C;
    __syncthreads();

    float vs = 0.0f;
    for (int c = t; c < C; c += 128) {
        const float d = xr[c] - mean;
        vs += d * d;
    }
    red[t] = vs;
    __syncthreads();
    for (int off = 64; off > 0; off >>= 1) {
        if (t < off) red[t] += red[t + off];
        __syncthreads();
    }
    const float rstd = rsqrtf(red[0] / (float)C + 1e-5f);
    __syncthreads();

    for (int c = t; c < C; c += 128)
        orow[c] = (bf16)((xr[c] - mean) * rstd * gw[c] + bw[c]);
}

// ---------------------------------------------------------------------------
// Pair-bias attention core (H=16, D=24) — tiny FLOPs, plain VALU.
// grid (256 i, 16 h), block 256 (thread j).
// ---------------------------------------------------------------------------
__global__ void __launch_bounds__(256)
pb_attn_kernel(const float* __restrict__ q, const float* __restrict__ k,
               const float* __restrict__ v, const float* __restrict__ g,
               const float* __restrict__ bias, float* __restrict__ o) {
    const int i = blockIdx.x, h = blockIdx.y, j = threadIdx.x;
    __shared__ float sh[256];
    __shared__ float red[256];
    const float scale = 0.20412414523193154f;  // 1/sqrt(24)

    float acc = 0.0f;
    for (int d = 0; d < 24; ++d)
        acc += q[(long)i * CS + h * 24 + d] * k[(long)j * CS + h * 24 + d];
    const float logit = acc * scale + bias[((long)i * 256 + j) * 16 + h];

    red[j] = logit;
    __syncthreads();
    for (int off = 128; off > 0; off >>= 1) {
        if (j < off) red[j] = fmaxf(red[j], red[j + off]);
        __syncthreads();
    }
    const float mx = red[0];
    __syncthreads();

    const float e = __expf(logit - mx);
    sh[j] = e;
    red[j] = e;
    __syncthreads();
    for (int off = 128; off > 0; off >>= 1) {
        if (j < off) red[j] += red[j + off];
        __syncthreads();
    }
    const float invs = 1.0f / red[0];
    __syncthreads();

    if (j < 24) {
        float sum = 0.0f;
        for (int jj = 0; jj < 256; ++jj)
            sum += sh[jj] * v[(long)jj * CS + h * 24 + j];
        o[(long)i * CS + h * 24 + j] = sum * invs * g[(long)i * CS + h * 24 + j];
    }
}

// ---------------------------------------------------------------------------
// Elementwise helpers.
// ---------------------------------------------------------------------------
__global__ void cvt_bf16_kernel(const float* __restrict__ a, bf16* __restrict__ o,
                                size_t n) {
    const size_t i = (size_t)blockIdx.x * blockDim.x + threadIdx.x;
    if (i < n) o[i] = (bf16)a[i];
}

// Weight convert+transpose to NT layout: o[n*K+k] = w[k*N+n], rows >= N zeroed.
__global__ void cvt_w_nt_kernel(const float* __restrict__ w, bf16* __restrict__ o,
                                int K, int N, int Npad) {
    const size_t i = (size_t)blockIdx.x * blockDim.x + threadIdx.x;
    if (i >= (size_t)Npad * K) return;
    const int n = (int)(i / (size_t)K);
    const int k = (int)(i % (size_t)K);
    o[i] = (n < N) ? (bf16)w[(size_t)k * N + n] : (bf16)0.0f;
}

__global__ void ew_sigmul_kernel(const float* __restrict__ a, const float* __restrict__ b,
                                 bf16* __restrict__ o, size_t n) {
    const size_t i = (size_t)blockIdx.x * blockDim.x + threadIdx.x;
    if (i < n) o[i] = (bf16)(sigmoidf_(a[i]) * b[i]);
}

__global__ void ew_silumul_kernel(const float* __restrict__ a, const float* __restrict__ b,
                                  bf16* __restrict__ o, size_t n) {
    const size_t i = (size_t)blockIdx.x * blockDim.x + threadIdx.x;
    if (i < n) {
        const float x = a[i];
        o[i] = (bf16)(x * sigmoidf_(x) * b[i]);
    }
}

// out[d][x][y] = in[(x*256+y)*128 + d] (swap=0) or in[(y*256+x)*128 + d] (swap=1)
__global__ void transpose_dmajor_kernel(const bf16* __restrict__ in,
                                        bf16* __restrict__ out, int swap) {
    const size_t idx = (size_t)blockIdx.x * blockDim.x + threadIdx.x;
    if (idx >= (size_t)CZ * NN) return;
    const int d = (int)(idx >> 16);
    const int xy = (int)(idx & 65535);
    const int x = xy >> 8, y = xy & 255;
    const int r = swap ? (y * 256 + x) : (x * 256 + y);
    out[idx] = in[(long)r * CZ + d];
}

// ===========================================================================
// Host side.
// ===========================================================================
struct TMW { const float *ln_in_g, *ln_in_b, *ln_out_g, *ln_out_b;
             const bf16 *a1, *a2, *b1, *b2, *g, *out; };
struct TAW { const float *ln_g, *ln_b;
             const bf16 *wq, *wk, *wv, *wb, *wg, *wo; };
struct TRW { const float *ln_g, *ln_b; const bf16 *wa, *wb, *wo; };
struct PBW { const float *bq, *lna_g, *lna_b, *lnb_g, *lnb_b;
             const bf16 *wq, *wk, *wv, *wb, *wg, *wo; };
struct BlkW { PBW pb; TAW ta_end, ta_start; TMW tm_in, tm_out; TRW ts, tz; };

extern "C" void kernel_launch(void* const* d_in, const int* in_sizes, int n_in,
                              void* d_out, int out_size, void* d_ws, size_t ws_size,
                              hipStream_t stream) {
    (void)in_sizes; (void)n_in; (void)out_size; (void)ws_size;

    const float* s_in = (const float*)d_in[0];
    const float* z_in = (const float*)d_in[1];

    // ---- workspace carve (~209 MB) ----------------------------------------
    char* wsp = (char*)d_ws;
    auto carve = [&](size_t bytes) -> char* {
        char* p = wsp;
        wsp += (bytes + 255) & ~(size_t)255;
        return p;
    };
    float* Z   = (float*)carve((size_t)NN * CZ * 4);       // z working copy
    float* SWk = (float*)carve((size_t)NRES * CS * 4);     // s working copy
    bf16*  ZN  = (bf16*) carve((size_t)NN * CZ * 2);       // LN(z) / generic bf16
    bf16*  ZNS = (bf16*) carve((size_t)NRES * CS * 2);     // LN(s)
    bf16*  BFS = (bf16*) carve((size_t)NRES * CS * 2);     // small bf16 temp
    bf16*  BF1 = (bf16*) carve((size_t)NN * CZ * 2);
    bf16*  BF2 = (bf16*) carve((size_t)NN * CZ * 2);
    bf16*  BF3 = (bf16*) carve((size_t)NN * CZ * 2);
    bf16*  BF4 = (bf16*) carve((size_t)NN * CZ * 2);
    float* F1  = (float*)carve((size_t)NN * CZ * 4);
    float* F2  = (float*)carve((size_t)NN * CZ * 4);
    float* SM  = (float*)carve((size_t)NN * 16 * 4);       // attn bias buffers
    float* PQ  = (float*)carve((size_t)NRES * CS * 4);
    float* PK  = (float*)carve((size_t)NRES * CS * 4);
    float* PV  = (float*)carve((size_t)NRES * CS * 4);
    float* PG  = (float*)carve((size_t)NRES * CS * 4);
    float* PO  = (float*)carve((size_t)NRES * CS * 4);
    char*  WB  = carve((size_t)16 * 1024 * 1024);          // bf16 NT weights

    hipMemcpyAsync(Z, z_in, (size_t)NN * CZ * 4, hipMemcpyDeviceToDevice, stream);
    hipMemcpyAsync(SWk, s_in, (size_t)NRES * CS * 4, hipMemcpyDeviceToDevice, stream);

    // ---- parameter walk (pytree leaf order: dict keys sorted) -------------
    int pi = 2;
    char* wbcur = WB;
    auto nextf = [&]() -> const float* { return (const float*)d_in[pi++]; };
    // Convert weight [K,N] f32 -> NT bf16 [Npad][K], zero-padded rows.
    auto nextw = [&](int K, int N, int Npad) -> const bf16* {
        const float* src = nextf();
        bf16* dst = (bf16*)wbcur;
        const size_t n = (size_t)Npad * K;
        wbcur += (n * 2 + 255) & ~(size_t)255;
        cvt_w_nt_kernel<<<(unsigned)((n + 255) / 256), 256, 0, stream>>>(src, dst,
                                                                         K, N, Npad);
        return dst;
    };

    BlkW blocks[2];
    for (int b = 0; b < 2; ++b) {
        BlkW& B = blocks[b];
        // pb: bq, lna_b, lna_g, lnb_b, lnb_g, wb, wg, wk, wo, wq, wv
        B.pb.bq = nextf(); B.pb.lna_b = nextf(); B.pb.lna_g = nextf();
        B.pb.lnb_b = nextf(); B.pb.lnb_g = nextf();
        B.pb.wb = nextw(CZ, 16, 16);
        B.pb.wg = nextw(CS, CS, CS);
        B.pb.wk = nextw(CS, CS, CS);
        B.pb.wo = nextw(CS, CS, CS);
        B.pb.wq = nextw(CS, CS, CS);
        B.pb.wv = nextw(CS, CS, CS);
        // ta_end / ta_start: ln_b, ln_g, wb, wg, wk, wo, wq, wv
        for (int t = 0; t < 2; ++t) {
            TAW& A = (t == 0) ? B.ta_end : B.ta_start;
            A.ln_b = nextf(); A.ln_g = nextf();
            A.wb = nextw(CZ, 4, 16);            // pad head dim 4 -> 16
            A.wg = nextw(CZ, CZ, CZ);
            A.wk = nextw(CZ, CZ, CZ);
            A.wo = nextw(CZ, CZ, CZ);
            A.wq = nextw(CZ, CZ, CZ);
            A.wv = nextw(CZ, CZ, CZ);
        }
        // tm_in / tm_out: a1,a2,b1,b2,g,ln_in_b,ln_in_g,ln_out_b,ln_out_g,out
        for (int t = 0; t < 2; ++t) {
            TMW& M = (t == 0) ? B.tm_in : B.tm_out;
            M.a1 = nextw(CZ, CZ, CZ);
            M.a2 = nextw(CZ, CZ, CZ);
            M.b1 = nextw(CZ, CZ, CZ);
            M.b2 = nextw(CZ, CZ, CZ);
            M.g  = nextw(CZ, CZ, CZ);
            M.ln_in_b = nextf(); M.ln_in_g = nextf();
            M.ln_out_b = nextf(); M.ln_out_g = nextf();
            M.out = nextw(CZ, CZ, CZ);
        }
        // ts: ln_b, ln_g, wa, wb, wo  (c = 384)
        B.ts.ln_b = nextf(); B.ts.ln_g = nextf();
        B.ts.wa = nextw(CS, 4 * CS, 4 * CS);
        B.ts.wb = nextw(CS, 4 * CS, 4 * CS);
        B.ts.wo = nextw(4 * CS, CS, CS);
        // tz: ln_b, ln_g, wa, wb, wo  (c = 128)
        B.tz.ln_b = nextf(); B.tz.ln_g = nextf();
        B.tz.wa = nextw(CZ, 4 * CZ, 4 * CZ);
        B.tz.wb = nextw(CZ, 4 * CZ, 4 * CZ);
        B.tz.wo = nextw(4 * CZ, CZ, CZ);
    }

    // ---- launch helpers ----------------------------------------------------
    auto gemm = [&](const bf16* A, const bf16* Wt, const float* bias, const float* res,
                    const float* gate, float* out, bf16* outb,
                    int M, int N, int K, int mode) {
        dim3 g((M + 127) / 128, (N + 63) / 64);
        gemm_bf16_kernel<<<g, 256, 0, stream>>>(A, Wt, bias, res, gate, out, outb,
                                                M, N, K, mode);
    };
    auto gemm16 = [&](const bf16* A, const bf16* Wt, float* out, int M, int K) {
        gemm_n16_kernel<<<(M + 127) / 128, 256, 0, stream>>>(A, Wt, out, M, K);
    };
    auto lnorm = [&](const float* x, const float* gw, const float* bw, bf16* o,
                     int R, int C) {
        ln_kernel<<<R, 128, 0, stream>>>(x, gw, bw, o, C);
    };
    auto cvt = [&](const float* a, bf16* o, size_t n) {
        cvt_bf16_kernel<<<(unsigned)((n + 255) / 256), 256, 0, stream>>>(a, o, n);
    };
    auto sigmul = [&](const float* a, const float* b, bf16* o, size_t n) {
        ew_sigmul_kernel<<<(unsigned)((n + 255) / 256), 256, 0, stream>>>(a, b, o, n);
    };
    auto silumul = [&](const float* a, const float* b, bf16* o, size_t n) {
        ew_silumul_kernel<<<(unsigned)((n + 255) / 256), 256, 0, stream>>>(a, b, o, n);
    };

    auto run_tri_mul = [&](const TMW& w, int outgoing) {
        lnorm(Z, w.ln_in_g, w.ln_in_b, ZN, NN, CZ);
        gemm(ZN, w.a1, nullptr, nullptr, nullptr, F1, nullptr, NN, CZ, CZ, 0);
        gemm(ZN, w.a2, nullptr, nullptr, nullptr, F2, nullptr, NN, CZ, CZ, 0);
        sigmul(F1, F2, BF1, (size_t)NN * CZ);                       // a
        gemm(ZN, w.b1, nullptr, nullptr, nullptr, F1, nullptr, NN, CZ, CZ, 0);
        gemm(ZN, w.b2, nullptr, nullptr, nullptr, F2, nullptr, NN, CZ, CZ, 0);
        sigmul(F1, F2, BF2, (size_t)NN * CZ);                       // b
        gemm(ZN, w.g, nullptr, nullptr, nullptr, F1, nullptr, NN, CZ, CZ, 0);
        const int swap = outgoing ? 0 : 1;
        transpose_dmajor_kernel<<<32768, 256, 0, stream>>>(BF1, BF3, swap);
        transpose_dmajor_kernel<<<32768, 256, 0, stream>>>(BF2, BF4, swap);
        tri_einsum_kernel<<<dim3(4, 4, 128), 128, 0, stream>>>(BF3, BF4, F2);
        lnorm(F2, w.ln_out_g, w.ln_out_b, ZN, NN, CZ);
        gemm(ZN, w.out, nullptr, Z, F1, Z, nullptr, NN, CZ, CZ, 2); // z += sig(g)*proj
    };

    auto run_tri_attn = [&](const TAW& w, int starting) {
        lnorm(Z, w.ln_g, w.ln_b, ZN, NN, CZ);
        gemm(ZN, w.wq, nullptr, nullptr, nullptr, nullptr, BF1, NN, CZ, CZ, 0);
        gemm(ZN, w.wk, nullptr, nullptr, nullptr, nullptr, BF2, NN, CZ, CZ, 0);
        gemm(ZN, w.wv, nullptr, nullptr, nullptr, nullptr, BF3, NN, CZ, CZ, 0);
        gemm16(ZN, w.wb, SM, NN, CZ);                                // bias [nn,16]
        gemm(ZN, w.wg, nullptr, nullptr, nullptr, F1, nullptr, NN, CZ, CZ, 3);
        tri_attn_kernel<<<dim3(256, 8, 4), 64, 0, stream>>>(BF1, BF2, BF3, SM, F1,
                                                            BF4, starting);
        gemm(BF4, w.wo, nullptr, Z, nullptr, Z, nullptr, NN, CZ, CZ, 1); // z += o@wo
    };

    auto run_transition = [&](const TRW& w, float* X, int R, int c, bf16* lnbuf) {
        lnorm(X, w.ln_g, w.ln_b, lnbuf, R, c);
        const int CH = (R > 16384) ? 16384 : R;
        for (int r0 = 0; r0 < R; r0 += CH) {
            const bf16* Ab = lnbuf + (size_t)r0 * c;
            gemm(Ab, w.wa, nullptr, nullptr, nullptr, F1, nullptr, CH, 4 * c, c, 0);
            gemm(Ab, w.wb, nullptr, nullptr, nullptr, F2, nullptr, CH, 4 * c, c, 0);
            silumul(F1, F2, BF1, (size_t)CH * 4 * c);
            gemm(BF1, w.wo, nullptr, X + (size_t)r0 * c, nullptr,
                 X + (size_t)r0 * c, nullptr, CH, c, 4 * c, 1);
        }
    };

    auto run_pb = [&](const PBW& w) {
        lnorm(SWk, w.lna_g, w.lna_b, ZNS, NRES, CS);
        gemm(ZNS, w.wq, w.bq, nullptr, nullptr, PQ, nullptr, NRES, CS, CS, 0);
        gemm(ZNS, w.wk, nullptr, nullptr, nullptr, PK, nullptr, NRES, CS, CS, 0);
        gemm(ZNS, w.wv, nullptr, nullptr, nullptr, PV, nullptr, NRES, CS, CS, 0);
        gemm(ZNS, w.wg, nullptr, nullptr, nullptr, PG, nullptr, NRES, CS, CS, 3);
        lnorm(Z, w.lnb_g, w.lnb_b, ZN, NN, CZ);
        gemm16(ZN, w.wb, SM, NN, CZ);                                // bias [nn,16]
        pb_attn_kernel<<<dim3(256, 16), 256, 0, stream>>>(PQ, PK, PV, PG, SM, PO);
        cvt(PO, BFS, (size_t)NRES * CS);
        gemm(BFS, w.wo, nullptr, SWk, nullptr, SWk, nullptr, NRES, CS, CS, 1);
    };

    // ---- forward -----------------------------------------------------------
    for (int b = 0; b < 2; ++b) {
        const BlkW& P = blocks[b];
        run_tri_mul(P.tm_out, 1);
        run_tri_mul(P.tm_in, 0);
        run_tri_attn(P.ta_start, 1);
        run_tri_attn(P.ta_end, 0);
        run_transition(P.tz, Z, NN, CZ, ZN);
        run_pb(P.pb);
        run_transition(P.ts, SWk, NRES, CS, ZNS);
    }

    // ---- outputs: (s, z) concatenated f32 ----------------------------------
    float* out = (float*)d_out;
    hipMemcpyAsync(out, SWk, (size_t)NRES * CS * 4, hipMemcpyDeviceToDevice, stream);
    hipMemcpyAsync(out + (size_t)NRES * CS, Z, (size_t)NN * CZ * 4,
                   hipMemcpyDeviceToDevice, stream);
}
